// Detector_54365696032903
// MI455X (gfx1250) — compile-verified, matrix operand order checked by reading, not run on hardware
//
#include <hip/hip_runtime.h>
#include <hip/hip_bf16.h>
#include <math.h>

typedef __bf16 bf16_t;
typedef __attribute__((ext_vector_type(16))) __bf16 v16bf;
typedef __attribute__((ext_vector_type(8)))  __bf16 v8bf;
typedef __attribute__((ext_vector_type(8)))  float  v8f;

// ---------------- problem constants ----------------
constexpr int Bc = 4, Lc = 24, Tc = 400, Dc = 1024, Hc = 512;
constexpr int Gc = 4, Sc = 6, HEADSc = 2, Dhc = 256;
constexpr int NSTREAM = Bc * Lc;          // 96 expert streams
constexpr int NALL    = NSTREAM + Bc;     // + 4 shared-mean ("time_ctx") streams
constexpr int TPAD    = 416;              // 400 padded to multiple of 32 for K-loops
constexpr float TEMP = 1.75f, ALPHA = 0.5f, EPSf = 1e-5f;

// ---------------- device helpers ----------------
__device__ inline float geluf(float v) {
    return 0.5f * v * (1.f + erff(v * 0.70710678118654752f));
}
__device__ inline float wave_sum(float s) {
    #pragma unroll
    for (int m = 16; m >= 1; m >>= 1) s += __shfl_xor(s, m, 32);
    return s;
}
__device__ inline float wave_max(float s) {
    #pragma unroll
    for (int m = 16; m >= 1; m >>= 1) s = fmaxf(s, __shfl_xor(s, m, 32));
    return s;
}
__device__ inline float block_sum(float v, float* buf, int nw) {
    v = wave_sum(v);
    int w = threadIdx.x >> 5, ln = threadIdx.x & 31;
    __syncthreads();
    if (ln == 0) buf[w] = v;
    __syncthreads();
    float r = 0.f;
    for (int i = 0; i < nw; ++i) r += buf[i];
    return r;
}

// ---- CDNA5 async global->LDS copy (ASYNCcnt-tracked, ISA §15.18.3 op 98) ----
// Dynamic LDS starts at offset 0 in kernels with no static __shared__, so a
// plain byte offset is a valid LDS address for the VDST operand.
__device__ inline void async_copy_b128(unsigned lds_off, const void* gsrc) {
    asm volatile("global_load_async_to_lds_b128 %0, %1, off"
                 :: "v"(lds_off), "v"(gsrc) : "memory");
}
__device__ inline void wait_async() {
    asm volatile("s_wait_asynccnt 0x0" ::: "memory");
}

// ---- WMMA fragment loaders (16x16x32 bf16, wave32 layouts per CDNA5 ISA §7.12.2) ----
__device__ inline v16bf cat8(v8bf lo, v8bf hi) {
    return __builtin_shufflevector(lo, hi, 0, 1, 2, 3, 4, 5, 6, 7,
                                   8, 9, 10, 11, 12, 13, 14, 15);
}
// A: 16x32 (MxK) row-major. Lane reads two contiguous 8-element runs:
// regs 0-3 -> K = kk+0..7 ; regs 4-7 -> K = kk+16..23  (kk = (lane>=16)*8)
__device__ inline v16bf load_a_frag(const bf16_t* src, int ld) {
    int l = threadIdx.x & 31;
    int m  = l & 15;
    int kk = (l >> 4) * 8;
    const bf16_t* p = src + m * ld + kk;
    return cat8(*(const v8bf*)p, *(const v8bf*)(p + 16));
}
// B = Mᵀ where M is [N][K] row-major (Q·Kᵀ, P·Vᵀ-layout): B[k][n] = M[n][k].
__device__ inline v16bf load_bt_frag(const bf16_t* src, int ld) {
    int l = threadIdx.x & 31;
    int n  = l & 15;
    int kk = (l >> 4) * 8;
    const bf16_t* p = src + n * ld + kk;
    return cat8(*(const v8bf*)p, *(const v8bf*)(p + 16));
}
// Packed weights: one contiguous 32-byte chunk per lane per 32x16 tile.
__device__ inline v16bf load_b_packed(const bf16_t* base, int tile) {
    return ((const v16bf*)base)[(size_t)tile * 32 + (threadIdx.x & 31)];
}
__device__ inline v8f wmma_bf16(v16bf a, v16bf b, v8f c) {
    return __builtin_amdgcn_wmma_f32_16x16x32_bf16(false, a, false, b, (short)0, c,
                                                   false, false);
}

// ---------------- weight preparation ----------------
// Pack row-major W[K][N] (optionally row-scaled by grow[k], i.e. LN gamma folded)
// into WMMA B-fragment order: out[tile][lane][16 bf16], tile = kt*(N/16)+ntile.
__global__ void pack_b(const float* __restrict__ w, const float* __restrict__ grow,
                       bf16_t* __restrict__ out, int K, int N) {
    int idx = blockIdx.x * blockDim.x + threadIdx.x;
    int Ntl = N >> 4;
    int total = (K >> 5) * Ntl * 32;
    if (idx >= total) return;
    int lane = idx & 31;
    int tile = idx >> 5;
    int ntile = tile % Ntl, kt = tile / Ntl;
    int n  = ntile * 16 + (lane & 15);
    int kk = (lane >> 4) * 8;
    v16bf vv;
    #pragma unroll
    for (int j = 0; j < 8; ++j) {
        int k = kt * 32 + ((j < 4) ? (kk + 2 * j) : (kk + 16 + 2 * (j - 4)));
        float g0 = grow ? grow[k] : 1.f;
        float g1 = grow ? grow[k + 1] : 1.f;
        vv[2 * j]     = (bf16_t)(w[(size_t)k * N + n] * g0);
        vv[2 * j + 1] = (bf16_t)(w[(size_t)(k + 1) * N + n] * g1);
    }
    ((v16bf*)out)[(size_t)tile * 32 + lane] = vv;
}
// out[c] = b[c] + sum_r lnb[r]*w[r,c]  (LN beta folded through the GEMM)
__global__ void fold_bias(const float* __restrict__ w, const float* __restrict__ lnb,
                          const float* __restrict__ b, float* __restrict__ out,
                          int rows, int cols) {
    int c = blockIdx.x * blockDim.x + threadIdx.x;
    if (c >= cols) return;
    float a = b[c];
    for (int r = 0; r < rows; ++r) a += lnb[r] * w[(size_t)r * cols + c];
    out[c] = a;
}

// ---------------- K2: fused shared+group MLP ----------------
// grid: (B*L)*25 blocks, 256 thr. LDS: xhat[16][1024] + hs/hg[16][512] bf16 = 64 KB
__global__ void mlp_fused(const float* __restrict__ x,
                          const bf16_t* __restrict__ W1s, const bf16_t* __restrict__ W1g,
                          const bf16_t* __restrict__ W2s, const bf16_t* __restrict__ W2g,
                          const float* __restrict__ b1s, const float* __restrict__ b1g,
                          const float* __restrict__ s_b2, const float* __restrict__ g_b2,
                          const float* __restrict__ mlp_alpha,
                          bf16_t* __restrict__ weighted, float* __restrict__ shared_mean) {
    extern __shared__ char smem[];
    bf16_t* xh = (bf16_t*)smem;                       // [16][1024]
    bf16_t* hs = (bf16_t*)(smem + 32768);             // [16][512]
    bf16_t* hg = (bf16_t*)(smem + 32768 + 16384);     // [16][512]

    int blk = blockIdx.x;
    int tt = blk % 25;
    int bl = blk / 25;
    int b = bl / Lc, l = bl % Lc, grp = l / Sc;
    int t0 = tt * 16;
    float alpha = mlp_alpha[l];
    const float* xbase = x + ((size_t)bl * Tc) * Dc;

    int w = threadIdx.x >> 5, ln = threadIdx.x & 31;

    // LayerNorm-hat (shared across shared/group paths); 8 waves x 2 rows,
    // each lane owns a contiguous 32-float block -> float4 loads, b128 stores.
    for (int rr = 0; rr < 2; ++rr) {
        int row = w * 2 + rr;
        const float4* xr4 = (const float4*)(xbase + (size_t)(t0 + row) * Dc) + ln * 8;
        float v[32];
        float s = 0.f;
        #pragma unroll
        for (int i = 0; i < 8; ++i) {
            float4 f = xr4[i];
            v[4 * i] = f.x; v[4 * i + 1] = f.y; v[4 * i + 2] = f.z; v[4 * i + 3] = f.w;
            s += f.x + f.y + f.z + f.w;
        }
        s = wave_sum(s);
        float mu = s * (1.f / Dc), q = 0.f;
        #pragma unroll
        for (int i = 0; i < 32; ++i) { float d = v[i] - mu; q += d * d; }
        q = wave_sum(q);
        float rstd = rsqrtf(q * (1.f / Dc) + EPSf);
        #pragma unroll
        for (int c = 0; c < 4; ++c) {
            v8bf pv;
            #pragma unroll
            for (int j = 0; j < 8; ++j) pv[j] = (bf16_t)((v[c * 8 + j] - mu) * rstd);
            *(v8bf*)(xh + row * Dc + ln * 32 + c * 8) = pv;
        }
    }
    __syncthreads();

    const bf16_t* W1gp = W1g + (size_t)grp * Dc * Hc;
    constexpr int Nt1 = Hc / 16;   // 32 tile-columns
    // GEMM1: [16,1024] @ [1024,512]; shared & group paths share the A fragment
    for (int nt = 0; nt < 4; ++nt) {
        int ncol = w * 4 + nt;
        int n0 = ncol * 16;
        v8f accs = {}, accg = {};
        for (int kt = 0; kt < Dc / 32; ++kt) {
            v16bf a  = load_a_frag(xh + kt * 32, Dc);
            v16bf bs = load_b_packed(W1s,  kt * Nt1 + ncol);
            v16bf bg = load_b_packed(W1gp, kt * Nt1 + ncol);
            if (kt + 1 < Dc / 32)
                __builtin_prefetch((const v16bf*)W1s + (size_t)(kt * Nt1 + Nt1 + ncol) * 32,
                                   0, 1);
            accs = wmma_bf16(a, bs, accs);
            accg = wmma_bf16(a, bg, accg);
        }
        int n = n0 + (ln & 15), mhi = (ln >> 4) * 8;
        #pragma unroll
        for (int r = 0; r < 8; ++r) {
            int m = mhi + r;
            hs[m * Hc + n] = (bf16_t)geluf(accs[r] + b1s[n]);
            hg[m * Hc + n] = (bf16_t)geluf(accg[r] + b1g[grp * Hc + n]);
        }
    }
    __syncthreads();

    const bf16_t* W2gp = W2g + (size_t)grp * Hc * Hc;
    // GEMM2: [16,512] @ [512,512]; weighted = shared + alpha*group
    for (int nt = 0; nt < 4; ++nt) {
        int ncol = w * 4 + nt;
        int n0 = ncol * 16;
        v8f accs = {}, accg = {};
        for (int kt = 0; kt < Hc / 32; ++kt) {
            v16bf as = load_a_frag(hs + kt * 32, Hc);
            v16bf ag = load_a_frag(hg + kt * 32, Hc);
            v16bf bs = load_b_packed(W2s,  kt * Nt1 + ncol);
            v16bf bg = load_b_packed(W2gp, kt * Nt1 + ncol);
            accs = wmma_bf16(as, bs, accs);
            accg = wmma_bf16(ag, bg, accg);
        }
        int n = n0 + (ln & 15), mhi = (ln >> 4) * 8;
        #pragma unroll
        for (int r = 0; r < 8; ++r) {
            int t = t0 + mhi + r;
            float sh = accs[r] + s_b2[n];
            float go = accg[r] + g_b2[grp * Hc + n];
            weighted[((size_t)bl * Tc + t) * Hc + n] = (bf16_t)(sh + alpha * go);
            atomicAdd(&shared_mean[((size_t)b * Tc + t) * Hc + n], sh * (1.f / Lc));
        }
    }
}

// ---------------- K3: pool-LN (folded) + QKV projection ----------------
// grid: NALL*25, 256 thr. LDS: what[16][512] bf16 = 16 KB
__global__ void qkv_kernel(const bf16_t* __restrict__ weighted,
                           const float* __restrict__ shared_mean,
                           const bf16_t* __restrict__ Pin, const float* __restrict__ bqkv,
                           bf16_t* __restrict__ qb, bf16_t* __restrict__ kb,
                           bf16_t* __restrict__ vt) {
    extern __shared__ char smem[];
    bf16_t* wh = (bf16_t*)smem;   // [16][512]
    int blk = blockIdx.x;
    int tt = blk % 25, s = blk / 25;
    int t0 = tt * 16;
    int w = threadIdx.x >> 5, ln = threadIdx.x & 31;

    // row-wise LN-hat; each lane owns a contiguous 16-element block
    for (int rr = 0; rr < 2; ++rr) {
        int row = w * 2 + rr, t = t0 + row;
        float v[16], sm = 0.f;
        if (s < NSTREAM) {
            const bf16_t* wr = weighted + ((size_t)s * Tc + t) * Hc + ln * 16;
            v8bf lo = *(const v8bf*)wr, hi = *(const v8bf*)(wr + 8);
            #pragma unroll
            for (int i = 0; i < 8; ++i) { v[i] = (float)lo[i]; v[8 + i] = (float)hi[i]; }
        } else {
            const float4* sr = (const float4*)(shared_mean +
                               ((size_t)(s - NSTREAM) * Tc + t) * Hc) + ln * 4;
            #pragma unroll
            for (int i = 0; i < 4; ++i) {
                float4 f = sr[i];
                v[4 * i] = f.x; v[4 * i + 1] = f.y; v[4 * i + 2] = f.z; v[4 * i + 3] = f.w;
            }
        }
        #pragma unroll
        for (int i = 0; i < 16; ++i) sm += v[i];
        sm = wave_sum(sm);
        float mu = sm * (1.f / Hc), q = 0.f;
        #pragma unroll
        for (int i = 0; i < 16; ++i) { float d = v[i] - mu; q += d * d; }
        q = wave_sum(q);
        float rstd = rsqrtf(q * (1.f / Hc) + EPSf);
        #pragma unroll
        for (int c = 0; c < 2; ++c) {
            v8bf pv;
            #pragma unroll
            for (int j = 0; j < 8; ++j) pv[j] = (bf16_t)((v[c * 8 + j] - mu) * rstd);
            *(v8bf*)(wh + row * Hc + ln * 16 + c * 8) = pv;
        }
    }
    __syncthreads();

    // [16,512] @ [512,1536]; 12 tile-columns per wave
    constexpr int NtQ = (3 * Hc) / 16;   // 96
    for (int j = 0; j < 12; ++j) {
        int ncol = w * 12 + j;
        int n0 = ncol * 16;
        v8f acc = {};
        for (int kt = 0; kt < Hc / 32; ++kt) {
            v16bf a = load_a_frag(wh + kt * 32, Hc);
            v16bf b = load_b_packed(Pin, kt * NtQ + ncol);
            acc = wmma_bf16(a, b, acc);
        }
        int col = n0 + (ln & 15), mhi = (ln >> 4) * 8;
        int sect = col >> 9, rem = col & 511, head = rem >> 8, dh = rem & 255;
        size_t shd = (size_t)(s * HEADSc + head);
        if (sect == 2) {
            // V stored transposed [sh][Dh][TPAD]: one contiguous b128 store per lane
            v8bf pv;
            #pragma unroll
            for (int r = 0; r < 8; ++r) pv[r] = (bf16_t)(acc[r] + bqkv[col]);
            *(v8bf*)(vt + (shd * Dhc + dh) * TPAD + t0 + mhi) = pv;
        } else {
            #pragma unroll
            for (int r = 0; r < 8; ++r) {
                int t = t0 + mhi + r;
                float val = acc[r] + bqkv[col];
                size_t idx = (shd * TPAD + t) * Dhc + dh;
                if (sect == 0) qb[idx] = (bf16_t)(val * 0.0625f);  // fold 1/sqrt(Dh)
                else           kb[idx] = (bf16_t)val;
            }
        }
    }
}

// ---------------- K4: fused attention (scores + softmax + P·V) ----------------
// grid: (NALL*HEADS)*25, 256 thr. LDS: q[16][256] + S[16][416]f32 + P[16][416]bf16
__global__ void attn_kernel(const bf16_t* __restrict__ qb, const bf16_t* __restrict__ kb,
                            const bf16_t* __restrict__ vt, bf16_t* __restrict__ attn_out) {
    extern __shared__ char smem[];
    bf16_t* qt = (bf16_t*)smem;                       // [16][256], LDS offset 0
    float*  Sr = (float*)(smem + 8192);               // [16][416]
    bf16_t* P  = (bf16_t*)(smem + 8192 + 26624);      // [16][416]
    int blk = blockIdx.x;
    int qtile = blk % 25, sh = blk / 25;              // sh = stream*2 + head
    int t0 = qtile * 16;
    int tid = threadIdx.x, w = tid >> 5, ln = tid & 31;

    // async DMA the contiguous 8KB Q tile into LDS (ASYNCcnt path)
    {
        const char* gsrc = (const char*)(qb + ((size_t)sh * TPAD + t0) * Dhc);
        for (int i = tid; i < 512; i += 256)
            async_copy_b128((unsigned)(i * 16), gsrc + (size_t)i * 16);
        wait_async();
    }
    __syncthreads();

    // scores S[16,400] = Q · Kᵀ
    const bf16_t* Kb = kb + (size_t)sh * TPAD * Dhc;
    for (int nt = w; nt < 25; nt += 8) {
        int n0 = nt * 16;
        v8f acc = {};
        for (int k0 = 0; k0 < Dhc; k0 += 32) {
            v16bf a = load_a_frag(qt + k0, Dhc);
            v16bf b = load_bt_frag(Kb + (size_t)n0 * Dhc + k0, Dhc);
            acc = wmma_bf16(a, b, acc);
        }
        int n = n0 + (ln & 15), mhi = (ln >> 4) * 8;
        #pragma unroll
        for (int r = 0; r < 8; ++r) Sr[(mhi + r) * TPAD + n] = acc[r];
    }
    __syncthreads();

    // softmax over full 400-wide rows; 8 waves x 2 rows
    for (int rr = 0; rr < 2; ++rr) {
        int row = w * 2 + rr;
        float e[13], mx = -1e30f;
        #pragma unroll
        for (int i = 0; i < 13; ++i) {
            int c = ln + 32 * i;
            e[i] = (c < Tc) ? Sr[row * TPAD + c] : -1e30f;
            mx = fmaxf(mx, e[i]);
        }
        mx = wave_max(mx);
        float sum = 0.f;
        #pragma unroll
        for (int i = 0; i < 13; ++i) {
            int c = ln + 32 * i;
            float ex = (c < Tc) ? __expf(e[i] - mx) : 0.f;
            e[i] = ex; sum += ex;
        }
        sum = wave_sum(sum);
        float inv = 1.f / sum;
        #pragma unroll
        for (int i = 0; i < 13; ++i) {
            int c = ln + 32 * i;
            P[row * TPAD + c] = (bf16_t)(e[i] * inv);   // padding cols get 0
        }
    }
    __syncthreads();

    // O[16,256] = P[16,416] @ V[416,256]; V is transposed -> contiguous B loads
    const bf16_t* Vt = vt + (size_t)sh * Dhc * TPAD;
    int stream = sh >> 1, head = sh & 1;
    for (int j = 0; j < 2; ++j) {
        int n0 = (w * 2 + j) * 16;
        v8f acc = {};
        for (int k0 = 0; k0 < TPAD; k0 += 32) {
            v16bf a = load_a_frag(P + k0, TPAD);
            v16bf b = load_bt_frag(Vt + (size_t)n0 * TPAD + k0, TPAD);
            acc = wmma_bf16(a, b, acc);
        }
        int col = head * Dhc + n0 + (ln & 15), mhi = (ln >> 4) * 8;
        #pragma unroll
        for (int r = 0; r < 8; ++r) {
            int t = t0 + mhi + r;
            attn_out[((size_t)stream * Tc + t) * Hc + col] = (bf16_t)acc[r];
        }
    }
}

// ---------------- K5: out-proj GEMM fused with mean over T ----------------
// grid: NALL blocks, 256 thr. LDS: atile[16][512] bf16 = 16 KB (LDS offset 0)
__global__ void outproj_pool(const bf16_t* __restrict__ attn_out,
                             const bf16_t* __restrict__ Pout,
                             const float* __restrict__ p_out_b,
                             float* __restrict__ out_tp, float* __restrict__ time_ctx) {
    extern __shared__ char smem[];
    bf16_t* at = (bf16_t*)smem;     // [16][512]
    int s = blockIdx.x;
    int tid = threadIdx.x, w = tid >> 5, ln = tid & 31;
    constexpr int Nt1 = Hc / 16;
    float csum[4] = {0.f, 0.f, 0.f, 0.f};

    for (int qtile = 0; qtile < 25; ++qtile) {
        int t0 = qtile * 16;
        // async DMA the contiguous 16KB tile into LDS
        const char* gsrc = (const char*)(attn_out + ((size_t)s * Tc + t0) * Hc);
        for (int i = tid; i < 1024; i += 256)
            async_copy_b128((unsigned)(i * 16), gsrc + (size_t)i * 16);
        wait_async();
        __syncthreads();
        for (int nt = 0; nt < 4; ++nt) {
            int ncol = w * 4 + nt;
            v8f acc = {};
            for (int kt = 0; kt < Hc / 32; ++kt) {
                v16bf a = load_a_frag(at + kt * 32, Hc);
                v16bf b = load_b_packed(Pout, kt * Nt1 + ncol);
                acc = wmma_bf16(a, b, acc);
            }
            float cs = 0.f;
            #pragma unroll
            for (int r = 0; r < 8; ++r) cs += acc[r];   // partial column sum (8 rows)
            csum[nt] += cs;
        }
        __syncthreads();
    }
    #pragma unroll
    for (int nt = 0; nt < 4; ++nt) csum[nt] += __shfl_xor(csum[nt], 16, 32);
    if (ln < 16) {
        #pragma unroll
        for (int nt = 0; nt < 4; ++nt) {
            int col = w * 64 + nt * 16 + ln;
            float tp = csum[nt] * (1.f / Tc) + p_out_b[col];
            if (s < NSTREAM) out_tp[(size_t)s * Hc + col] = tp;
            else             time_ctx[(size_t)(s - NSTREAM) * Hc + col] = tp;
        }
    }
}

// ---------------- K6: router ----------------
__global__ void router_kernel(const float* __restrict__ tp,
                              const float* __restrict__ r_w1, const float* __restrict__ r_b1,
                              const float* __restrict__ r_w2, const float* __restrict__ r_b2,
                              float* __restrict__ routing) {
    int bl = blockIdx.x, t = threadIdx.x;   // 128 threads
    __shared__ float buf[4];
    const float* v = tp + (size_t)bl * Hc;
    float acc = r_b1[t];
    for (int d = 0; d < Hc; ++d) acc += v[d] * r_w1[d * 128 + t];
    float part = geluf(acc) * r_w2[t];
    float logit = block_sum(part, buf, 4) + r_b2[0];
    if (t == 0) {
        float sig = 1.f / (1.f + __expf(-logit / TEMP));
        routing[bl] = ALPHA * sig + (1.f - ALPHA) / (float)Lc;
    }
}

// ---------------- K7: soft fusion + post-LN + mean over T ----------------
__global__ void fusion_kernel(const bf16_t* __restrict__ weighted,
                              const float* __restrict__ routing,
                              const float* __restrict__ pn_g, const float* __restrict__ pn_b,
                              float* __restrict__ fusion_acc) {
    int bt = blockIdx.x;
    int b = bt / Tc, t = bt % Tc;
    int tid = threadIdx.x;
    int h0 = 2 * tid, h1 = 2 * tid + 1;
    float f0 = 0.f, f1 = 0.f;
    for (int l = 0; l < Lc; ++l) {
        float rw = routing[b * Lc + l];
        const bf16_t* wr = weighted + ((size_t)(b * Lc + l) * Tc + t) * Hc;
        f0 += rw * (float)wr[h0];
        f1 += rw * (float)wr[h1];
    }
    __shared__ float buf[8];
    float s = block_sum(f0 + f1, buf, 8);
    float q = block_sum(f0 * f0 + f1 * f1, buf, 8);
    float mu = s * (1.f / Hc);
    float rstd = rsqrtf(q * (1.f / Hc) - mu * mu + EPSf);
    float y0 = (f0 - mu) * rstd * pn_g[h0] + pn_b[h0];
    float y1 = (f1 - mu) * rstd * pn_g[h1] + pn_b[h1];
    atomicAdd(&fusion_acc[b * Hc + h0], y0 * (1.f / Tc));
    atomicAdd(&fusion_acc[b * Hc + h1], y1 * (1.f / Tc));
}

// ---------------- K8: fused_all = LN(fusion) + LN(time_ctx) ----------------
__global__ void final_fuse_kernel(const float* __restrict__ fusion_acc,
                                  const float* __restrict__ time_ctx,
                                  const float* __restrict__ pnf_g,
                                  const float* __restrict__ pnf_b,
                                  float* __restrict__ fused_out) {
    int b = blockIdx.x, tid = threadIdx.x;
    int h0 = tid, h1 = tid + 256;
    const float* fa = fusion_acc + (size_t)b * Hc;
    const float* tc = time_ctx + (size_t)b * Hc;
    float a0 = fa[h0], a1 = fa[h1], c0 = tc[h0], c1 = tc[h1];
    __shared__ float buf[8];
    float sa = block_sum(a0 + a1, buf, 8);
    float qa = block_sum(a0 * a0 + a1 * a1, buf, 8);
    float sc = block_sum(c0 + c1, buf, 8);
    float qc = block_sum(c0 * c0 + c1 * c1, buf, 8);
    float mua = sa * (1.f / Hc), muc = sc * (1.f / Hc);
    float ra = rsqrtf(qa * (1.f / Hc) - mua * mua + EPSf);
    float rc = rsqrtf(qc * (1.f / Hc) - muc * muc + EPSf);
    fused_out[b * Hc + h0] = ((a0 - mua) * ra + (c0 - muc) * rc) * pnf_g[h0]
                             + 2.f * pnf_b[h0];
    fused_out[b * Hc + h1] = ((a1 - mua) * ra + (c1 - muc) * rc) * pnf_g[h1]
                             + 2.f * pnf_b[h1];
}

// ---------------- K9: classifier ----------------
__global__ void classifier_kernel(const float* __restrict__ fused,
                                  const float* __restrict__ c_w1, const float* __restrict__ c_b1,
                                  const float* __restrict__ c_w2, const float* __restrict__ c_b2,
                                  const float* __restrict__ c_w3, const float* __restrict__ c_b3,
                                  float* __restrict__ logits) {
    int b = blockIdx.x, t = threadIdx.x;   // 256 threads
    __shared__ float h1[256];
    __shared__ float h2[128];
    const float* fv = fused + (size_t)b * Hc;
    float a = c_b1[t];
    for (int d = 0; d < Hc; ++d) a += fv[d] * c_w1[d * 256 + t];
    h1[t] = geluf(a);
    __syncthreads();
    if (t < 128) {
        float a2 = c_b2[t];
        for (int d = 0; d < 256; ++d) a2 += h1[d] * c_w2[d * 128 + t];
        h2[t] = geluf(a2);
    }
    __syncthreads();
    if (t < 2) {
        float a3 = c_b3[t];
        for (int d = 0; d < 128; ++d) a3 += h2[d] * c_w3[d * 2 + t];
        logits[b * 2 + t] = a3;
    }
}

// ---------------- host launcher ----------------
extern "C" void kernel_launch(void* const* d_in, const int* in_sizes, int n_in,
                              void* d_out, int out_size, void* d_ws, size_t ws_size,
                              hipStream_t stream) {
    const float* x        = (const float*)d_in[0];
    const float* mlp_alpha= (const float*)d_in[1];
    const float* s_ln_g   = (const float*)d_in[2];
    const float* s_ln_b   = (const float*)d_in[3];
    const float* s_w1     = (const float*)d_in[4];
    const float* s_b1     = (const float*)d_in[5];
    const float* s_w2     = (const float*)d_in[6];
    const float* s_b2     = (const float*)d_in[7];
    const float* g_ln_g   = (const float*)d_in[8];
    const float* g_ln_b   = (const float*)d_in[9];
    const float* g_w1     = (const float*)d_in[10];
    const float* g_b1     = (const float*)d_in[11];
    const float* g_w2     = (const float*)d_in[12];
    const float* g_b2     = (const float*)d_in[13];
    const float* p_ln_g   = (const float*)d_in[14];
    const float* p_ln_b   = (const float*)d_in[15];
    const float* p_in_w   = (const float*)d_in[16];
    const float* p_in_b   = (const float*)d_in[17];
    const float* p_out_w  = (const float*)d_in[18];
    const float* p_out_b  = (const float*)d_in[19];
    const float* r_w1     = (const float*)d_in[20];
    const float* r_b1     = (const float*)d_in[21];
    const float* r_w2     = (const float*)d_in[22];
    const float* r_b2     = (const float*)d_in[23];
    const float* pn_g     = (const float*)d_in[24];
    const float* pn_b     = (const float*)d_in[25];
    const float* pnf_g    = (const float*)d_in[26];
    const float* pnf_b    = (const float*)d_in[27];
    const float* c_w1     = (const float*)d_in[28];
    const float* c_b1     = (const float*)d_in[29];
    const float* c_w2     = (const float*)d_in[30];
    const float* c_b2     = (const float*)d_in[31];
    const float* c_w3     = (const float*)d_in[32];
    const float* c_b3     = (const float*)d_in[33];

    // output regions (fp32): logits[8] | routing[96] | fused_all[2048] | time_pooled[49152]
    float* out_logits  = (float*)d_out;
    float* out_routing = out_logits + 8;
    float* out_fused   = out_routing + 96;
    float* out_tp      = out_fused + Bc * Hc;

    // workspace carve-up
    char* ws = (char*)d_ws;
    size_t cur = 0;
    auto alloc = [&](size_t bytes) -> char* {
        char* p = ws + cur;
        cur += (bytes + 255) & ~(size_t)255;
        return p;
    };
    bf16_t* W1s   = (bf16_t*)alloc((size_t)Dc * Hc * 2);
    bf16_t* W1g   = (bf16_t*)alloc((size_t)Gc * Dc * Hc * 2);
    bf16_t* W2s   = (bf16_t*)alloc((size_t)Hc * Hc * 2);
    bf16_t* W2g   = (bf16_t*)alloc((size_t)Gc * Hc * Hc * 2);
    bf16_t* Pin   = (bf16_t*)alloc((size_t)Hc * 3 * Hc * 2);
    bf16_t* Pout  = (bf16_t*)alloc((size_t)Hc * Hc * 2);
    float*  b1s   = (float*)alloc((size_t)Hc * 4);
    float*  b1g   = (float*)alloc((size_t)Gc * Hc * 4);
    float*  bqkv  = (float*)alloc((size_t)3 * Hc * 4);
    bf16_t* weighted    = (bf16_t*)alloc((size_t)NSTREAM * Tc * Hc * 2);
    float*  shared_mean = (float*)alloc((size_t)Bc * Tc * Hc * 4);
    size_t qkv_elems = (size_t)NALL * HEADSc * TPAD * Dhc;
    bf16_t* qb = (bf16_t*)alloc(qkv_elems * 2);
    bf16_t* kb = (bf16_t*)alloc(qkv_elems * 2);
    bf16_t* vt = (bf16_t*)alloc(qkv_elems * 2);          // transposed [sh][Dh][TPAD]
    bf16_t* attn_out   = (bf16_t*)alloc((size_t)NALL * Tc * Hc * 2);
    float*  time_ctx   = (float*)alloc((size_t)Bc * Hc * 4);
    float*  fusion_acc = (float*)alloc((size_t)Bc * Hc * 4);

    // zero accumulators + V (padding cols t=400..415 must be 0 for the P·V K-loop)
    hipMemsetAsync(shared_mean, 0, (size_t)Bc * Tc * Hc * 4, stream);
    hipMemsetAsync(vt, 0, qkv_elems * 2, stream);
    hipMemsetAsync(fusion_acc, 0, (size_t)Bc * Hc * 4, stream);

    // pack weights into WMMA fragment order (LN gamma folded into GEMM-1 / QKV)
    auto pk = [&](const float* w, const float* g, bf16_t* o, int K, int N) {
        int total = (K / 32) * (N / 16) * 32;
        pack_b<<<(total + 255) / 256, 256, 0, stream>>>(w, g, o, K, N);
    };
    pk(s_w1, s_ln_g, W1s, Dc, Hc);
    for (int g = 0; g < Gc; ++g)
        pk(g_w1 + (size_t)g * Dc * Hc, g_ln_g + (size_t)g * Dc,
           W1g + (size_t)g * Dc * Hc, Dc, Hc);
    pk(s_w2, nullptr, W2s, Hc, Hc);
    for (int g = 0; g < Gc; ++g)
        pk(g_w2 + (size_t)g * Hc * Hc, nullptr, W2g + (size_t)g * Hc * Hc, Hc, Hc);
    pk(p_in_w, p_ln_g, Pin, Hc, 3 * Hc);
    pk(p_out_w, nullptr, Pout, Hc, Hc);

    fold_bias<<<(Hc + 255) / 256, 256, 0, stream>>>(s_w1, s_ln_b, s_b1, b1s, Dc, Hc);
    for (int g = 0; g < Gc; ++g)
        fold_bias<<<(Hc + 255) / 256, 256, 0, stream>>>(
            g_w1 + (size_t)g * Dc * Hc, g_ln_b + (size_t)g * Dc,
            g_b1 + (size_t)g * Hc, b1g + (size_t)g * Hc, Dc, Hc);
    fold_bias<<<(3 * Hc + 255) / 256, 256, 0, stream>>>(p_in_w, p_ln_b, p_in_b, bqkv,
                                                        Hc, 3 * Hc);

    // main pipeline
    mlp_fused<<<NSTREAM * 25, 256, 65536, stream>>>(
        x, W1s, W1g, W2s, W2g, b1s, b1g, s_b2, g_b2, mlp_alpha,
        weighted, shared_mean);

    qkv_kernel<<<NALL * 25, 256, 16384, stream>>>(
        weighted, shared_mean, Pin, bqkv, qb, kb, vt);

    attn_kernel<<<NALL * HEADSc * 25, 256, 48128, stream>>>(qb, kb, vt, attn_out);

    outproj_pool<<<NALL, 256, 16384, stream>>>(attn_out, Pout, p_out_b,
                                               out_tp, time_ctx);

    router_kernel<<<NSTREAM, 128, 0, stream>>>(out_tp, r_w1, r_b1, r_w2, r_b2,
                                               out_routing);

    fusion_kernel<<<Bc * Tc, 256, 0, stream>>>(weighted, out_routing, pn_g, pn_b,
                                               fusion_acc);

    final_fuse_kernel<<<Bc, 256, 0, stream>>>(fusion_acc, time_ctx, pnf_g, pnf_b,
                                              out_fused);

    classifier_kernel<<<Bc, 256, 0, stream>>>(out_fused, c_w1, c_b1, c_w2, c_b2,
                                              c_w3, c_b3, out_logits);
}
